// TimeMixing_62680752718588
// MI455X (gfx1250) — matrix operand order repaired and checked by reading, not compiled, not run on hardware
//
#include <hip/hip_runtime.h>
#include <stdint.h>

#define B_ 4
#define T_ 4096
#define C_ 1024
#define M_TOTAL (B_*T_)     // 16384
#define SEG_ 32             // WKV segments
#define LSEG_ (T_/SEG_)     // 128 steps per segment

typedef __attribute__((ext_vector_type(16))) __bf16 v16bf;
typedef __attribute__((ext_vector_type(8)))  __bf16 bf16x8;
typedef __attribute__((ext_vector_type(4)))  __bf16 bf16x4;
typedef __attribute__((ext_vector_type(4)))  float  f32x4;
typedef __attribute__((ext_vector_type(8)))  float  v8f;

// Low 32 bits of a generic pointer to __shared__ = LDS byte offset
__device__ __forceinline__ uint32_t lds_addr32(const void* p) {
    return (uint32_t)(uintptr_t)p;
}
// ASYNCcnt-tracked 16B copy global->LDS (GVS mode: SGPR64 base + VGPR32 offset).
__device__ __forceinline__ void async_load_b128(uint32_t lds, const __bf16* sbase,
                                                uint32_t voff_bytes) {
    asm volatile("global_load_async_to_lds_b128 %0, %1, %2"
                 :: "v"(lds), "v"(voff_bytes), "s"(sbase) : "memory");
}
__device__ __forceinline__ void wait_async0() {
    asm volatile("s_wait_asynccnt 0x0" ::: "memory");
}

// ---------------- f32 weights -> bf16 (vectorized x4) ----------------------
__global__ __launch_bounds__(256)
void wcvt_kernel(const float* __restrict__ w, __bf16* __restrict__ o, int n4) {
    int i = blockIdx.x * 256 + threadIdx.x;
    if (i < n4) {
        f32x4 f = *(const f32x4*)(w + (size_t)i * 4);
        bf16x4 b;
        b.x = (__bf16)f.x; b.y = (__bf16)f.y; b.z = (__bf16)f.z; b.w = (__bf16)f.w;
        *(bf16x4*)(o + (size_t)i * 4) = b;
    }
}

// ---------------- time-mix: xk/xv/xr = x*mu + prev*(1-mu) ------------------
__global__ __launch_bounds__(256)
void mix_kernel(const float* __restrict__ x,
                const float* __restrict__ mk, const float* __restrict__ mv,
                const float* __restrict__ mr,
                __bf16* __restrict__ xk, __bf16* __restrict__ xv,
                __bf16* __restrict__ xr)
{
    int i = blockIdx.x * 256 + threadIdx.x;        // over B*T*C/4
    size_t e = (size_t)i * 4;
    int c4 = (int)(e & (C_ - 1));
    int t  = (int)((e >> 10) & (T_ - 1));
    f32x4 cur = *(const f32x4*)(x + e);
    f32x4 prv = (t > 0) ? *(const f32x4*)(x + e - C_) : (f32x4){0.f, 0.f, 0.f, 0.f};
    f32x4 a = *(const f32x4*)(mk + c4);
    f32x4 b = *(const f32x4*)(mv + c4);
    f32x4 r = *(const f32x4*)(mr + c4);
    bf16x4 ok, ov, orr;
    ok.x = (__bf16)(cur.x*a.x + prv.x*(1.f-a.x));  ok.y = (__bf16)(cur.y*a.y + prv.y*(1.f-a.y));
    ok.z = (__bf16)(cur.z*a.z + prv.z*(1.f-a.z));  ok.w = (__bf16)(cur.w*a.w + prv.w*(1.f-a.w));
    ov.x = (__bf16)(cur.x*b.x + prv.x*(1.f-b.x));  ov.y = (__bf16)(cur.y*b.y + prv.y*(1.f-b.y));
    ov.z = (__bf16)(cur.z*b.z + prv.z*(1.f-b.z));  ov.w = (__bf16)(cur.w*b.w + prv.w*(1.f-b.w));
    orr.x = (__bf16)(cur.x*r.x + prv.x*(1.f-r.x)); orr.y = (__bf16)(cur.y*r.y + prv.y*(1.f-r.y));
    orr.z = (__bf16)(cur.z*r.z + prv.z*(1.f-r.z)); orr.w = (__bf16)(cur.w*r.w + prv.w*(1.f-r.w));
    *(bf16x4*)(xk + e) = ok;
    *(bf16x4*)(xv + e) = ov;
    *(bf16x4*)(xr + e) = orr;
}

// ---------------- GEMM: out[m,n] = sum_k A[m,k]*W[n,k] ---------------------
// Block tile 128x128, BK=32, double-buffered LDS fed by async-to-LDS copies.
// LDS rows padded to 40 elements (80B) -> conflict-free 16B fragment reads.
// Inner loop unrolled x2 with statically-selected buffers (branch-free).
#define BM 128
#define BN 128
#define BK 32
#define LDST 40

__device__ __forceinline__ void issue_tiles(uint32_t ldsA, uint32_t ldsB,
        const __bf16* Abase, const __bf16* Wbase,
        uint32_t dA0, uint32_t dA1, uint32_t gA0, uint32_t gA1, uint32_t ko)
{
    async_load_b128(ldsA + dA0, Abase, gA0 + ko);
    async_load_b128(ldsA + dA1, Abase, gA1 + ko);
    async_load_b128(ldsB + dA0, Wbase, gA0 + ko);
    async_load_b128(ldsB + dA1, Wbase, gA1 + ko);
}

__device__ __forceinline__ void compute_step(const __bf16* __restrict__ sAc,
                                             const __bf16* __restrict__ sBc,
                                             int mW, int nW, int ln, int lh,
                                             v8f acc[2][4])
{
    v16bf afrag[2];
#pragma unroll
    for (int mt = 0; mt < 2; ++mt) {
        const __bf16* ap = sAc + (mW + mt * 16 + ln) * LDST + lh * 8;
        bf16x8 lo = *(const bf16x8*)ap;          // K = b8..b8+7
        bf16x8 hi = *(const bf16x8*)(ap + 16);   // K = b8+16..b8+23
        afrag[mt] = __builtin_shufflevector(lo, hi,
            0, 1, 2, 3, 4, 5, 6, 7, 8, 9, 10, 11, 12, 13, 14, 15);
    }
    v16bf bfrag[4];
#pragma unroll
    for (int nt = 0; nt < 4; ++nt) {
        const __bf16* bp = sBc + (nW + nt * 16 + ln) * LDST + lh * 8;
        bf16x8 lo = *(const bf16x8*)bp;          // K = lh*8 .. +7
        bf16x8 hi = *(const bf16x8*)(bp + 16);   // K = lh*8+16 .. +23
        bfrag[nt] = __builtin_shufflevector(lo, hi,
            0, 1, 2, 3, 4, 5, 6, 7, 8, 9, 10, 11, 12, 13, 14, 15);
    }
#pragma unroll
    for (int mt = 0; mt < 2; ++mt)
#pragma unroll
        for (int nt = 0; nt < 4; ++nt)
            acc[mt][nt] = __builtin_amdgcn_wmma_f32_16x16x32_bf16(
                false, afrag[mt], false, bfrag[nt],
                (short)0, acc[mt][nt], false, false);
}

__global__ __launch_bounds__(256)
void gemm_bf16_wmma(const __bf16* __restrict__ A, const __bf16* __restrict__ W,
                    float* __restrict__ out, int M, int N, int K)
{
    __shared__ __bf16 sA[2][BM * LDST];
    __shared__ __bf16 sB[2][BN * LDST];

    const int tid  = threadIdx.x;
    const int lane = tid & 31;
    const int wave = tid >> 5;             // 0..7
    const int ln   = lane & 15;
    const int lh   = lane >> 4;
    const int mW = (wave & 3) * 32;        // 4 waves along M
    const int nW = (wave >> 2) * 64;       // 2 waves along N
    const int mBlk = blockIdx.y * BM;
    const int nBlk = blockIdx.x * BN;

    // async copy: 512 16B chunks per tile, 2 per thread
    const int r0 = tid >> 2,         q0 = tid & 3;
    const int r1 = (tid + 256) >> 2, q1 = tid & 3;
    const uint32_t aL0 = lds_addr32(&sA[0][0]), aL1 = lds_addr32(&sA[1][0]);
    const uint32_t bL0 = lds_addr32(&sB[0][0]), bL1 = lds_addr32(&sB[1][0]);
    const uint32_t dA0 = (uint32_t)(r0 * LDST * 2 + q0 * 16);
    const uint32_t dA1 = (uint32_t)(r1 * LDST * 2 + q1 * 16);
    const __bf16* Abase = A + (size_t)mBlk * K;
    const __bf16* Wbase = W + (size_t)nBlk * K;
    const uint32_t gA0 = (uint32_t)((r0 * K + q0 * 8) * 2);
    const uint32_t gA1 = (uint32_t)((r1 * K + q1 * 8) * 2);
    const __bf16* sA0 = sA[0]; const __bf16* sA1 = sA[1];
    const __bf16* sB0 = sB[0]; const __bf16* sB1 = sB[1];

    v8f acc[2][4] = {};

    // prologue: stage step 0 into buf0
    issue_tiles(aL0, bL0, Abase, Wbase, dA0, dA1, gA0, gA1, 0);
    wait_async0();
    __syncthreads();

    // steady state: branch-free, buffers statically selected.
    // invariant at loop head: buf0 holds data for step kb.
#pragma unroll 1
    for (int kb = 0; kb + 2 * BK < K; kb += 2 * BK) {
        issue_tiles(aL1, bL1, Abase, Wbase, dA0, dA1, gA0, gA1,
                    (uint32_t)((kb + BK) * 2));
        compute_step(sA0, sB0, mW, nW, ln, lh, acc);
        wait_async0();
        __syncthreads();
        issue_tiles(aL0, bL0, Abase, Wbase, dA0, dA1, gA0, gA1,
                    (uint32_t)((kb + 2 * BK) * 2));
        compute_step(sA1, sB1, mW, nW, ln, lh, acc);
        wait_async0();
        __syncthreads();
    }
    // tail: steps K-2*BK (in buf0) and K-BK
    issue_tiles(aL1, bL1, Abase, Wbase, dA0, dA1, gA0, gA1,
                (uint32_t)((K - BK) * 2));
    compute_step(sA0, sB0, mW, nW, ln, lh, acc);
    wait_async0();
    __syncthreads();
    compute_step(sA1, sB1, mW, nW, ln, lh, acc);

#pragma unroll
    for (int mt = 0; mt < 2; ++mt)
#pragma unroll
        for (int nt = 0; nt < 4; ++nt) {
            const int n = nBlk + nW + nt * 16 + ln;
#pragma unroll
            for (int j = 0; j < 8; ++j) {
                const int m = mBlk + mW + mt * 16 + lh * 8 + j;
                out[(size_t)m * N + n] = acc[mt][nt][j];
            }
        }
}

// ---------------- WKV pass 1: per-segment partial state --------------------
// idx = s*(B*C) + b*C + c ; runs segment s from zero state (no outputs).
__global__ __launch_bounds__(256)
void wkv_seg_partial(const float* __restrict__ k, const float* __restrict__ v,
                     const float* __restrict__ time_decay,
                     float* __restrict__ sn, float* __restrict__ sd,
                     float* __restrict__ sm)
{
    int idx = blockIdx.x * 256 + threadIdx.x;     // over B*C*SEG
    int bc = idx & (B_ * C_ - 1);
    int s  = idx >> 12;
    int b  = bc >> 10;
    int c  = bc & (C_ - 1);
    const float td = -__expf(time_decay[c]);
    size_t base = (size_t)b * T_ * C_ + (size_t)(s * LSEG_) * C_ + c;
    float num = 0.f, den = 0.f, m = -1e38f;
    for (int t = 0; t < LSEG_; ++t) {
        size_t o = base + (size_t)t * C_;
        float ck = k[o], cv = v[o];
        float ms  = fmaxf(m + td, ck);
        float e1s = __expf(m + td - ms);
        float e2s = __expf(ck - ms);
        num = e1s * num + e2s * cv;
        den = e1s * den + e2s;
        m = ms;
    }
    sn[idx] = num; sd[idx] = den; sm[idx] = m;
}

// ---------------- WKV pass 2: exclusive prefix of segment states -----------
// state_after_seg = decay^L (entry) + partial ; overwrite arrays with ENTRY state.
__global__ __launch_bounds__(256)
void wkv_seg_prefix(const float* __restrict__ time_decay,
                    float* __restrict__ sn, float* __restrict__ sd,
                    float* __restrict__ sm)
{
    int idx = blockIdx.x * 256 + threadIdx.x;     // over B*C
    int c = idx & (C_ - 1);
    const float td  = -__expf(time_decay[c]);
    const float Ltd = (float)LSEG_ * td;
    float num = 0.f, den = 0.f, m = -1e38f;
    for (int s = 0; s < SEG_; ++s) {
        int o = s * (B_ * C_) + idx;
        float pn = sn[o], pd = sd[o], pm = sm[o];
        sn[o] = num; sd[o] = den; sm[o] = m;      // entry state for segment s
        float me = m + Ltd;
        float mo = fmaxf(me, pm);
        float ea = __expf(me - mo);
        float eb = __expf(pm - mo);
        num = ea * num + eb * pn;
        den = ea * den + eb * pd;
        m = mo;
    }
}

// ---------------- WKV pass 3: replay segment, fuse sigmoid(r)*wkv -> bf16 --
__global__ __launch_bounds__(256)
void wkv_seg_out(const float* __restrict__ k, const float* __restrict__ v,
                 const float* __restrict__ r,
                 const float* __restrict__ time_first,
                 const float* __restrict__ time_decay,
                 const float* __restrict__ sn, const float* __restrict__ sd,
                 const float* __restrict__ sm,
                 __bf16* __restrict__ rwkv)
{
    int idx = blockIdx.x * 256 + threadIdx.x;     // over B*C*SEG
    int bc = idx & (B_ * C_ - 1);
    int s  = idx >> 12;
    int b  = bc >> 10;
    int c  = bc & (C_ - 1);
    const float td = -__expf(time_decay[c]);
    const float u  = time_first[c];
    float num = sn[idx], den = sd[idx], m = sm[idx];
    size_t base = (size_t)b * T_ * C_ + (size_t)(s * LSEG_) * C_ + c;
    for (int t = 0; t < LSEG_; ++t) {
        size_t o = base + (size_t)t * C_;
        float ck = k[o], cv = v[o];
        float mo = fmaxf(m, ck + u);
        float e1 = __expf(m - mo);
        float e2 = __expf(ck + u - mo);
        float wkv = (e1 * num + e2 * cv) / (e1 * den + e2);
        float sg  = 1.0f / (1.0f + __expf(-r[o]));
        rwkv[o] = (__bf16)(sg * wkv);
        float ms  = fmaxf(m + td, ck);
        float e1s = __expf(m + td - ms);
        float e2s = __expf(ck - ms);
        num = e1s * num + e2s * cv;
        den = e1s * den + e2s;
        m = ms;
    }
}

extern "C" void kernel_launch(void* const* d_in, const int* in_sizes, int n_in,
                              void* d_out, int out_size, void* d_ws, size_t ws_size,
                              hipStream_t stream)
{
    const float* x   = (const float*)d_in[0];
    const float* w_k = (const float*)d_in[1];
    const float* w_v = (const float*)d_in[2];
    const float* w_r = (const float*)d_in[3];
    const float* w_o = (const float*)d_in[4];
    const float* td  = (const float*)d_in[5];
    const float* tf  = (const float*)d_in[6];
    const float* mk  = (const float*)d_in[7];
    const float* mv  = (const float*)d_in[8];
    const float* mr  = (const float*)d_in[9];

    char* ws = (char*)d_ws;
    const size_t CC   = (size_t)C_ * C_;
    const size_t BTC  = (size_t)B_ * T_ * C_;
    const size_t SEGN = (size_t)B_ * C_ * SEG_;   // 131072

    __bf16* wkb = (__bf16*)ws;
    __bf16* wvb = wkb + CC;
    __bf16* wrb = wvb + CC;
    __bf16* wob = wrb + CC;
    __bf16* xk  = (__bf16*)(ws + 4 * CC * 2);
    __bf16* xv  = xk + BTC;
    __bf16* xr  = xv + BTC;
    float*  kf  = (float*)(ws + 4 * CC * 2 + 3 * BTC * 2);
    float*  vf  = kf + BTC;
    float*  rf  = vf + BTC;
    float*  sn  = rf + BTC;                       // segment state arrays
    float*  sd  = sn + SEGN;
    float*  sm  = sd + SEGN;
    __bf16* rwkv = xr;                            // reuse xr region

    const int wgrid = (int)((CC / 4 + 255) / 256);
    wcvt_kernel<<<wgrid, 256, 0, stream>>>(w_k, wkb, (int)(CC / 4));
    wcvt_kernel<<<wgrid, 256, 0, stream>>>(w_v, wvb, (int)(CC / 4));
    wcvt_kernel<<<wgrid, 256, 0, stream>>>(w_r, wrb, (int)(CC / 4));
    wcvt_kernel<<<wgrid, 256, 0, stream>>>(w_o, wob, (int)(CC / 4));

    mix_kernel<<<(int)(BTC / 4 / 256), 256, 0, stream>>>(x, mk, mv, mr, xk, xv, xr);

    dim3 gg(C_ / BN, M_TOTAL / BM);               // (8, 128)
    gemm_bf16_wmma<<<gg, 256, 0, stream>>>(xk, wkb, kf, M_TOTAL, C_, C_);
    gemm_bf16_wmma<<<gg, 256, 0, stream>>>(xv, wvb, vf, M_TOTAL, C_, C_);
    gemm_bf16_wmma<<<gg, 256, 0, stream>>>(xr, wrb, rf, M_TOTAL, C_, C_);

    wkv_seg_partial<<<(int)(SEGN / 256), 256, 0, stream>>>(kf, vf, td, sn, sd, sm);
    wkv_seg_prefix<<<(B_ * C_) / 256, 256, 0, stream>>>(td, sn, sd, sm);
    wkv_seg_out<<<(int)(SEGN / 256), 256, 0, stream>>>(kf, vf, rf, tf, td,
                                                       sn, sd, sm, rwkv);

    gemm_bf16_wmma<<<gg, 256, 0, stream>>>(rwkv, wob, (float*)d_out, M_TOTAL, C_, C_);
}
